// MultiHeadAttention_32530082300504
// MI455X (gfx1250) — matrix operand order repaired and checked by reading, not compiled, not run on hardware
//
#include <hip/hip_runtime.h>

// ---------------------------------------------------------------------------
// MHA forward for B=2, S=2048, D=1024, H=16, DH=64 on gfx1250 (MI455X).
// bf16 WMMA (v_wmma_f32_16x16x32_bf16), fp32 accumulation, flash-attention
// causal softmax, async global->LDS staging (double buffered) for K tiles.
// ---------------------------------------------------------------------------

typedef __bf16 v16bf __attribute__((ext_vector_type(16)));
typedef float  v8f   __attribute__((ext_vector_type(8)));

#define BATCH   2
#define SEQ     2048
#define DMODEL  1024
#define NHEAD   16
#define DHEAD   64
#define ROWS    (BATCH * SEQ)          // 4096

#if defined(__has_builtin)
#if __has_builtin(__builtin_amdgcn_global_load_async_to_lds_b128)
#define HAVE_ASYNC_LDS 1
#endif
#endif

#if defined(HAVE_ASYNC_LDS)
// exact pointee type the builtin expects: int __attribute__((vector_size(16)))
typedef int v4i_vs __attribute__((vector_size(16)));
typedef __attribute__((address_space(1))) v4i_vs as1_v4i;   // global
typedef __attribute__((address_space(3))) v4i_vs as3_v4i;   // LDS
#if __has_builtin(__builtin_amdgcn_s_wait_asynccnt)
#define WAIT_ASYNC(n) __builtin_amdgcn_s_wait_asynccnt(n)
#else
#define WAIT_ASYNC(n) asm volatile("s_wait_asynccnt " #n ::: "memory")
#endif
#else
#define WAIT_ASYNC(n)
#endif

static __device__ __forceinline__ unsigned short f2bf(float f) {
  union { float f; unsigned int u; } x; x.f = f;
  unsigned int u = x.u;
  u += 0x7FFFu + ((u >> 16) & 1u);     // round-to-nearest-even
  return (unsigned short)(u >> 16);
}

static __device__ __forceinline__ float bf2f(unsigned short b) {
  union { unsigned int u; float f; } x; x.u = ((unsigned int)b) << 16;
  return x.f;
}

static __device__ __forceinline__ __bf16 bfbits(unsigned short b) {
  union { unsigned short u; __bf16 h; } x; x.u = b; return x.h;
}

// ---------------------------------------------------------------------------
// fp32 -> bf16 bits conversion
// ---------------------------------------------------------------------------
__global__ void mha_f32_to_bf16(const float* __restrict__ in,
                                unsigned short* __restrict__ out, int n) {
  int i = blockIdx.x * blockDim.x + threadIdx.x;
  if (i < n) out[i] = f2bf(in[i]);
}

// ---------------------------------------------------------------------------
// bf16 GEMM: C = A[M,K] x B[K,N], 64x64 block tile, 4 waves of 32x32.
// B tile staged TRANSPOSED in LDS so both A and B fragments are b128 loads.
// MODE 0: write bf16 with [B,H,S,DH] head remap (Q/K/V)
// MODE 1: write fp32 row-major + bias (final projection)
// ---------------------------------------------------------------------------
#define BM 64
#define BN 64
#define BK 32
#define AST 40   // A tile LDS row stride (elems): 80B rows, 16B aligned
#define BTT 40   // transposed B tile LDS row stride (elems)

template <int MODE>
__global__ __launch_bounds__(128) void mha_gemm_bf16(
    const unsigned short* __restrict__ A,
    const unsigned short* __restrict__ Bw,
    unsigned short* __restrict__ outBf,
    float* __restrict__ outF,
    const float* __restrict__ bias,
    int M, int N, int K) {
  __shared__ __align__(16) unsigned short As[BM * AST];
  __shared__ __align__(16) unsigned short BsT[BN * BTT];

  const int tid  = threadIdx.x;
  const int lane = tid & 31;
  const int wave = tid >> 5;
  const int half = lane >> 4;
  const int l16  = lane & 15;

  const int blockM = blockIdx.x * BM;
  const int blockN = blockIdx.y * BN;
  const int waveM  = (wave & 1) * 32;
  const int waveN  = (wave >> 1) * 32;

  v8f acc[2][2];
#pragma unroll
  for (int mi = 0; mi < 2; ++mi)
#pragma unroll
    for (int ni = 0; ni < 2; ++ni)
#pragma unroll
      for (int j = 0; j < 8; ++j) acc[mi][ni][j] = 0.0f;

  for (int k0 = 0; k0 < K; k0 += BK) {
    __syncthreads();
    // A tile 64x32 row-major (1024 dwords, 8 per thread)
#pragma unroll
    for (int i = 0; i < 8; ++i) {
      int u = tid + i * 128;
      int r = u >> 4;
      int c = (u & 15) * 2;
      unsigned int v = *(const unsigned int*)(A + (size_t)(blockM + r) * K + k0 + c);
      *(unsigned int*)&As[r * AST + c] = v;
    }
    // B tile 32x64 staged transposed: BsT[n][k]
#pragma unroll
    for (int i = 0; i < 8; ++i) {
      int u = tid + i * 128;
      int r = u >> 5;          // k
      int c = (u & 31) * 2;    // n
      unsigned int v = *(const unsigned int*)(Bw + (size_t)(k0 + r) * N + blockN + c);
      BsT[(c + 0) * BTT + r] = (unsigned short)(v & 0xFFFFu);
      BsT[(c + 1) * BTT + r] = (unsigned short)(v >> 16);
    }
    __syncthreads();

    v16bf afrag[2], bfrag[2];
#pragma unroll
    for (int s = 0; s < 2; ++s) {
      int row = waveM + s * 16 + l16;
      union { uint4 q[2]; v16bf v; } au;
      au.q[0] = *(const uint4*)&As[row * AST + half * 16];       // K = h*16 + 0..7
      au.q[1] = *(const uint4*)&As[row * AST + half * 16 + 8];   // K = h*16 + 8..15
      afrag[s] = au.v;
      int col = waveN + s * 16 + l16;
      union { uint4 q[2]; v16bf v; } bu;
      const uint4* bp = (const uint4*)&BsT[col * BTT + half * 16]; // K = h*16 + 0..15
      bu.q[0] = bp[0];
      bu.q[1] = bp[1];
      bfrag[s] = bu.v;
    }
#pragma unroll
    for (int mi = 0; mi < 2; ++mi)
#pragma unroll
      for (int ni = 0; ni < 2; ++ni)
        acc[mi][ni] = __builtin_amdgcn_wmma_f32_16x16x32_bf16(
            false, afrag[mi], false, bfrag[ni], (short)0, acc[mi][ni], false, false);
  }

  // C layout: VGPR j -> row j + 8*half, col = l16
#pragma unroll
  for (int mi = 0; mi < 2; ++mi) {
#pragma unroll
    for (int ni = 0; ni < 2; ++ni) {
#pragma unroll
      for (int j = 0; j < 8; ++j) {
        int r = blockM + waveM + mi * 16 + j + half * 8;
        int c = blockN + waveN + ni * 16 + l16;
        float v = acc[mi][ni][j];
        if (MODE == 0) {
          int b = r >> 11, s = r & (SEQ - 1);
          int h = c >> 6, dh = c & (DHEAD - 1);
          outBf[(((size_t)(b * NHEAD + h) * SEQ) + s) * DHEAD + dh] = f2bf(v);
        } else {
          outF[(size_t)r * N + c] = v + bias[c];
        }
      }
    }
  }
}

// ---------------------------------------------------------------------------
// Flash attention (causal). Block = one (b,h) x 64 query rows, 4 waves x 16.
// K tiles: async global->LDS, double buffered. V tiles: staged transposed.
// ---------------------------------------------------------------------------
#define KST 72   // Ks row stride (elems): 144B rows, 16B aligned
#define VST 40   // Vt row stride (elems): 80B rows, 16B aligned

__global__ __launch_bounds__(128) void mha_flash_attn(
    const unsigned short* __restrict__ Qh,   // [B*H, S, DH] bf16
    const unsigned short* __restrict__ Kh,
    const unsigned short* __restrict__ Vh,
    unsigned short* __restrict__ Ctx) {      // [B*S, D] bf16
  __shared__ __align__(16) unsigned short Ks[2][32 * KST];   // 9216 B
  __shared__ __align__(16) unsigned short Vt[DHEAD * VST];   // 5120 B
  __shared__ __align__(16) unsigned short Pl[4][16 * 32];    // 4096 B

  const int tid  = threadIdx.x;
  const int lane = tid & 31;
  const int wave = tid >> 5;
  const int half = lane >> 4;
  const int l16  = lane & 15;

  const int bh    = blockIdx.x;
  const int qbase = blockIdx.y * 64;
  const int qrow0 = qbase + wave * 16;

  const unsigned short* Qb = Qh + (size_t)bh * SEQ * DHEAD;
  const unsigned short* Kb = Kh + (size_t)bh * SEQ * DHEAD;
  const unsigned short* Vb = Vh + (size_t)bh * SEQ * DHEAD;

  // issue one K tile (32 rows x 64 bf16 = 4KB) into Ks[buf]; 16B per thread x2
  auto issueK = [&](int kb, int buf) {
#pragma unroll
    for (int t = 0; t < 2; ++t) {
      int cid = tid + t * 128;
      int row = cid >> 3;      // 8 x 16B chunks per row
      int cc  = cid & 7;
      unsigned short* g = const_cast<unsigned short*>(
          Kb + (size_t)(kb + row) * DHEAD + cc * 8);
      unsigned short* l = &Ks[buf][row * KST + cc * 8];
#if defined(HAVE_ASYNC_LDS)
      __builtin_amdgcn_global_load_async_to_lds_b128(
          (as1_v4i*)g, (as3_v4i*)l, 0, 0);
#else
      *(uint4*)l = *(const uint4*)g;
#endif
    }
  };

  // Q A-fragments, 1/sqrt(DH)=0.125 folded in (exact exponent shift in bf16)
  v16bf qf[2];
#pragma unroll
  for (int c = 0; c < 2; ++c) {
    const unsigned short* qp = Qb + (size_t)(qrow0 + l16) * DHEAD + c * 32;
#pragma unroll
    for (int i = 0; i < 16; ++i) {
      int kk = i + half * 16;
      qf[c][i] = bfbits(f2bf(bf2f(qp[kk]) * 0.125f));
    }
  }

  v8f ctx[4];
#pragma unroll
  for (int nt = 0; nt < 4; ++nt)
#pragma unroll
    for (int j = 0; j < 8; ++j) ctx[nt][j] = 0.0f;

  float rmax[8], rsum[8];
#pragma unroll
  for (int j = 0; j < 8; ++j) { rmax[j] = -1e30f; rsum[j] = 0.0f; }

  const int nIter = (qbase + 64) / 32;
  issueK(0, 0);

  for (int it = 0; it < nIter; ++it) {
    const int kb  = it * 32;
    const int buf = it & 1;

    __syncthreads();                       // prev iter done reading Ks[buf^1], Vt
    if (it + 1 < nIter) issueK(kb + 32, buf ^ 1);

    // stage V tile transposed: Vt[d][k]
#pragma unroll
    for (int i = 0; i < 8; ++i) {
      int u = tid + i * 128;
      int r = u >> 5;          // k
      int c = (u & 31) * 2;    // d
      unsigned int v = *(const unsigned int*)(Vb + (size_t)(kb + r) * DHEAD + c);
      Vt[(c + 0) * VST + r] = (unsigned short)(v & 0xFFFFu);
      Vt[(c + 1) * VST + r] = (unsigned short)(v >> 16);
    }

    if (it + 1 < nIter) { WAIT_ASYNC(2); } else { WAIT_ASYNC(0); }
    __syncthreads();                       // Ks[buf] + Vt visible to all waves

    // ---- scores: two 16x16 tiles ----
    v8f s[2];
#pragma unroll
    for (int t = 0; t < 2; ++t) {
      v8f st;
#pragma unroll
      for (int j = 0; j < 8; ++j) st[j] = 0.0f;
#pragma unroll
      for (int c = 0; c < 2; ++c) {
        union { uint4 q[2]; v16bf v; } kf;
        const uint4* kp =
            (const uint4*)&Ks[buf][(t * 16 + l16) * KST + c * 32 + half * 16];
        kf.q[0] = kp[0];
        kf.q[1] = kp[1];
        st = __builtin_amdgcn_wmma_f32_16x16x32_bf16(
            false, qf[c], false, kf.v, (short)0, st, false, false);
      }
      s[t] = st;
    }

    // ---- causal mask (scale already folded into Q) ----
    const int qrb = qrow0 + half * 8;
#pragma unroll
    for (int j = 0; j < 8; ++j) {
      int q = qrb + j;
      if ((kb + l16) > q)      s[0][j] = -1e30f;
      if ((kb + 16 + l16) > q) s[1][j] = -1e30f;
    }

    // ---- online softmax ----
    float tmax[8];
#pragma unroll
    for (int j = 0; j < 8; ++j) tmax[j] = fmaxf(s[0][j], s[1][j]);
#pragma unroll
    for (int off = 1; off < 16; off <<= 1)
#pragma unroll
      for (int j = 0; j < 8; ++j)
        tmax[j] = fmaxf(tmax[j], __shfl_xor(tmax[j], off, 32));

    float e0[8], e1[8], tsum[8], corr[8];
#pragma unroll
    for (int j = 0; j < 8; ++j) {
      float nm = fmaxf(rmax[j], tmax[j]);
      corr[j] = __expf(rmax[j] - nm);
      rmax[j] = nm;
      e0[j] = __expf(s[0][j] - nm);
      e1[j] = __expf(s[1][j] - nm);
      tsum[j] = e0[j] + e1[j];
#pragma unroll
      for (int nt = 0; nt < 4; ++nt) ctx[nt][j] *= corr[j];
    }
#pragma unroll
    for (int off = 1; off < 16; off <<= 1)
#pragma unroll
      for (int j = 0; j < 8; ++j)
        tsum[j] += __shfl_xor(tsum[j], off, 32);
#pragma unroll
    for (int j = 0; j < 8; ++j) rsum[j] = rsum[j] * corr[j] + tsum[j];

    // ---- P (C layout fp32) -> LDS -> bf16 A fragment ----
    __syncthreads();
#pragma unroll
    for (int j = 0; j < 8; ++j) {
      Pl[wave][(j + half * 8) * 32 + l16]      = f2bf(e0[j]);
      Pl[wave][(j + half * 8) * 32 + 16 + l16] = f2bf(e1[j]);
    }
    __syncthreads();
    v16bf pf;
    {
      union { uint4 q[2]; v16bf v; } pu;
      pu.q[0] = *(const uint4*)&Pl[wave][l16 * 32 + half * 16];
      pu.q[1] = *(const uint4*)&Pl[wave][l16 * 32 + half * 16 + 8];
      pf = pu.v;
    }

    // ---- ctx += P x V ----
#pragma unroll
    for (int nt = 0; nt < 4; ++nt) {
      union { uint4 q[2]; v16bf v; } vf;
      const uint4* vp = (const uint4*)&Vt[(nt * 16 + l16) * VST + half * 16];
      vf.q[0] = vp[0];
      vf.q[1] = vp[1];
      ctx[nt] = __builtin_amdgcn_wmma_f32_16x16x32_bf16(
          false, pf, false, vf.v, (short)0, ctx[nt], false, false);
    }
  }

  // ---- normalize + write bf16 ctx in [B, S, D] layout ----
  const int b = bh >> 4, h = bh & (NHEAD - 1);
#pragma unroll
  for (int nt = 0; nt < 4; ++nt) {
#pragma unroll
    for (int j = 0; j < 8; ++j) {
      float v = ctx[nt][j] / rsum[j];
      int r = qrow0 + j + half * 8;
      int col = h * DHEAD + nt * 16 + l16;
      Ctx[((size_t)(b * SEQ + r)) * DMODEL + col] = f2bf(v);
    }
  }
}

// ---------------------------------------------------------------------------
// launch
// ---------------------------------------------------------------------------
extern "C" void kernel_launch(void* const* d_in, const int* in_sizes, int n_in,
                              void* d_out, int out_size, void* d_ws, size_t ws_size,
                              hipStream_t stream) {
  const float* X  = (const float*)d_in[0];
  const float* Wq = (const float*)d_in[1];
  const float* Wk = (const float*)d_in[2];
  const float* Wv = (const float*)d_in[3];
  const float* Wo = (const float*)d_in[4];
  const float* bo = (const float*)d_in[5];
  float* out = (float*)d_out;

  unsigned short* ws = (unsigned short*)d_ws;
  size_t off = 0;
  unsigned short* Xb  = ws + off; off += (size_t)ROWS * DMODEL;
  unsigned short* Wqb = ws + off; off += (size_t)DMODEL * DMODEL;
  unsigned short* Wkb = ws + off; off += (size_t)DMODEL * DMODEL;
  unsigned short* Wvb = ws + off; off += (size_t)DMODEL * DMODEL;
  unsigned short* Wob = ws + off; off += (size_t)DMODEL * DMODEL;
  unsigned short* Qh  = ws + off; off += (size_t)BATCH * NHEAD * SEQ * DHEAD;
  unsigned short* Kh  = ws + off; off += (size_t)BATCH * NHEAD * SEQ * DHEAD;
  unsigned short* Vh  = ws + off; off += (size_t)BATCH * NHEAD * SEQ * DHEAD;
  unsigned short* Ctx = ws + off; off += (size_t)ROWS * DMODEL;

  const int nX = ROWS * DMODEL;
  const int nW = DMODEL * DMODEL;
  mha_f32_to_bf16<<<(nX + 255) / 256, 256, 0, stream>>>(X, Xb, nX);
  mha_f32_to_bf16<<<(nW + 255) / 256, 256, 0, stream>>>(Wq, Wqb, nW);
  mha_f32_to_bf16<<<(nW + 255) / 256, 256, 0, stream>>>(Wk, Wkb, nW);
  mha_f32_to_bf16<<<(nW + 255) / 256, 256, 0, stream>>>(Wv, Wvb, nW);
  mha_f32_to_bf16<<<(nW + 255) / 256, 256, 0, stream>>>(Wo, Wob, nW);

  dim3 gg(ROWS / BM, DMODEL / BN);
  dim3 gb(128);
  mha_gemm_bf16<0><<<gg, gb, 0, stream>>>(Xb, Wqb, Qh, nullptr, nullptr,
                                          ROWS, DMODEL, DMODEL);
  mha_gemm_bf16<0><<<gg, gb, 0, stream>>>(Xb, Wkb, Kh, nullptr, nullptr,
                                          ROWS, DMODEL, DMODEL);
  mha_gemm_bf16<0><<<gg, gb, 0, stream>>>(Xb, Wvb, Vh, nullptr, nullptr,
                                          ROWS, DMODEL, DMODEL);

  mha_flash_attn<<<dim3(BATCH * NHEAD, SEQ / 64), 128, 0, stream>>>(Qh, Kh, Vh, Ctx);

  mha_gemm_bf16<1><<<gg, gb, 0, stream>>>(Ctx, Wob, nullptr, out, bo,
                                          ROWS, DMODEL, DMODEL);
}
